// Encoder_42013370089564
// MI455X (gfx1250) — compile-verified, matrix operand order checked by reading, not run on hardware
//
#include <hip/hip_runtime.h>

// ---------------------------------------------------------------------------
// 2-layer tanh RNN for MI455X (gfx1250, wave32, WMMA).
// f16 operands + f32 accumulation via v_wmma_f32_16x16x32_f16.
// Double-buffered LDS tiles filled with GLOBAL_LOAD_ASYNC_TO_LDS_B128
// (ASYNCcnt pipeline); all B fragments batch-loaded, WMMAs issued
// back-to-back; fully constant tile/K addressing via templates.
// ---------------------------------------------------------------------------

typedef __attribute__((ext_vector_type(16))) _Float16 v16h;
typedef __attribute__((ext_vector_type(8)))  float    v8f;
typedef __attribute__((ext_vector_type(4)))  int      v4i;

#define LDS_PITCH 40   // halves per LDS row: 80B pitch, 16B aligned, conflict-free

#if defined(__HIP_DEVICE_COMPILE__) && \
    __has_builtin(__builtin_amdgcn_global_load_async_to_lds_b128)
#define HAVE_ASYNC_LDS 1
#else
#define HAVE_ASYNC_LDS 0
#endif

#define AS1 __attribute__((address_space(1)))
#define AS3 __attribute__((address_space(3)))

// Copy one 16-byte chunk global -> LDS (async when available).
__device__ __forceinline__ void copy16(const _Float16* __restrict__ g,
                                       _Float16* __restrict__ l) {
#if HAVE_ASYNC_LDS
  AS1 void* gp = (AS1 void*)(void*)g;
  AS3 void* lp = (AS3 void*)(void*)l;
  __builtin_amdgcn_global_load_async_to_lds_b128((AS1 v4i*)gp, (AS3 v4i*)lp, 0, 0);
#else
  *(uint4*)l = *(const uint4*)g;
#endif
}

__device__ __forceinline__ void waitAsync() {
#if HAVE_ASYNC_LDS
#if __has_builtin(__builtin_amdgcn_s_wait_asynccnt)
  __builtin_amdgcn_s_wait_asynccnt(0);
#else
  asm volatile("s_wait_asynccnt 0x0" ::: "memory");
#endif
#endif
}

// Load one 16x32 f16 WMMA fragment from an LDS tile row.
// Per ISA 7.12.2: lane half hi holds K = {hi*8 .. +7} then {16+hi*8 .. +7}:
// two contiguous 16-byte runs -> two ds_load_b128.
__device__ __forceinline__ v16h ldsFrag(const _Float16* rowBase, int hi8) {
  union { v16h v; uint4 q[2]; } f;
  f.q[0] = *(const uint4*)(rowBase + hi8);
  f.q[1] = *(const uint4*)(rowBase + 16 + hi8);
  return f.v;
}

// C = A(MxKK,f16) * B(KKxNN,f16)  [BT is B^T: NN x KK row-major]
//   (+ biasA[n]+biasB[n])  (+ addend[m][n])  (optional tanh)
// Outputs: f32 C, f16 C, extra f32 copy (final hidden state).
// BM x BN block tile, 256 threads = 8 waves, wave grid (BM/16) x (8/(BM/16)).
template<int BM, int BN, int NN, int KK>
__global__ __launch_bounds__(256)
void wmma_gemm_kernel(const _Float16* __restrict__ A,
                      const _Float16* __restrict__ BT,
                      const float* __restrict__ biasA,
                      const float* __restrict__ biasB,
                      const float* __restrict__ addend,
                      float*    __restrict__ Cf32,
                      _Float16* __restrict__ Cf16,
                      float*    __restrict__ lastF32,
                      int act_tanh)
{
  constexpr int ROWW = BM / 16;        // waves along M
  constexpr int COLW = 8 / ROWW;       // wave groups along N
  constexpr int WN   = BN / COLW;      // columns per wave
  constexpr int TN   = WN / 16;        // 16x16 tiles per wave
  constexpr int CA   = (BM * 4) / 256; // 16B chunks per thread, A tile
  constexpr int CB   = (BN * 4) / 256; // 16B chunks per thread, B tile
  static_assert(KK % 64 == 0, "K must be a multiple of 64");

  const int tid  = threadIdx.x;
  const int wave = tid >> 5;           // wave32
  const int lane = tid & 31;
  const int lr   = lane & 15;
  const int hi   = lane >> 4;
  const int hi8  = hi * 8;

  const int rowStrip = wave % ROWW;
  const int colGrp   = wave / ROWW;

  const int m0 = blockIdx.y * BM;
  const int n0 = blockIdx.x * BN;

  __shared__ alignas(16) _Float16 sA[2][BM * LDS_PITCH];
  __shared__ alignas(16) _Float16 sB[2][BN * LDS_PITCH];

  // Per-thread fixed chunk coordinates (compile-time trip counts, no branches).
  const _Float16* aSrc[CA]; _Float16* aDst[2][CA];
  const _Float16* bSrc[CB]; _Float16* bDst[2][CB];
#pragma unroll
  for (int i = 0; i < CA; ++i) {
    const int c = tid + i * 256, r = c >> 2, s = c & 3;
    aSrc[i]    = A + (size_t)(m0 + r) * KK + s * 8;
    aDst[0][i] = &sA[0][r * LDS_PITCH + s * 8];
    aDst[1][i] = &sA[1][r * LDS_PITCH + s * 8];
  }
#pragma unroll
  for (int i = 0; i < CB; ++i) {
    const int c = tid + i * 256, r = c >> 2, s = c & 3;
    bSrc[i]    = BT + (size_t)(n0 + r) * KK + s * 8;
    bDst[0][i] = &sB[0][r * LDS_PITCH + s * 8];
    bDst[1][i] = &sB[1][r * LDS_PITCH + s * 8];
  }

  v8f acc[TN];
#pragma unroll
  for (int t = 0; t < TN; ++t)
    acc[t] = (v8f){0.f,0.f,0.f,0.f,0.f,0.f,0.f,0.f};

  const _Float16* aRow[2] = { sA[0] + (rowStrip * 16 + lr) * LDS_PITCH,
                              sA[1] + (rowStrip * 16 + lr) * LDS_PITCH };
  const _Float16* bRow[2] = { sB[0] + (colGrp * WN + lr) * LDS_PITCH,
                              sB[1] + (colGrp * WN + lr) * LDS_PITCH };

  // Issue all async fills for one K-slab into LDS buffer `buf`.
  auto fill = [&](int buf, int kc) {
#pragma unroll
    for (int i = 0; i < CA; ++i) copy16(aSrc[i] + kc, aDst[buf][i]);
#pragma unroll
    for (int i = 0; i < CB; ++i) copy16(bSrc[i] + kc, bDst[buf][i]);
  };

  // Batch-load A + all B fragments, then fire WMMAs back-to-back.
  auto compute = [&](int buf) {
    const v16h a = ldsFrag(aRow[buf], hi8);
    v16h bf[TN];
#pragma unroll
    for (int t = 0; t < TN; ++t)
      bf[t] = ldsFrag(bRow[buf] + t * 16 * LDS_PITCH, hi8);
#pragma unroll
    for (int t = 0; t < TN; ++t)
      acc[t] = __builtin_amdgcn_wmma_f32_16x16x32_f16(
          false, a, false, bf[t], (short)0, acc[t], false, false);
  };

  // Software pipeline: fill(next) overlaps compute(current); barriers publish.
  fill(0, 0);
  waitAsync(); __syncthreads();
#pragma unroll 1
  for (int kc = 0; kc < KK - 64; kc += 64) {
    fill(1, kc + 32);
    compute(0);
    waitAsync(); __syncthreads();
    fill(0, kc + 64);
    compute(1);
    waitAsync(); __syncthreads();
  }
  fill(1, KK - 32);
  compute(0);
  waitAsync(); __syncthreads();
  compute(1);

  // Epilogue. C/D layout: lanes 0-15 -> M=v, N=lane ; lanes 16-31 -> M=v+8.
#pragma unroll
  for (int t = 0; t < TN; ++t) {
    const int ncol = n0 + colGrp * WN + t * 16 + lr;
    float bias = 0.f;
    if (biasA) bias = biasA[ncol] + biasB[ncol];
#pragma unroll
    for (int v = 0; v < 8; ++v) {
      const int mrow = m0 + rowStrip * 16 + v + hi * 8;
      const size_t idx = (size_t)mrow * NN + ncol;
      float val = acc[t][v] + bias;
      if (addend)   val += addend[idx];
      if (act_tanh) val = tanhf(val);
      if (Cf32)    Cf32[idx]    = val;
      if (Cf16)    Cf16[idx]    = (_Float16)val;
      if (lastF32) lastF32[idx] = val;
    }
  }
}

// --- helpers ---------------------------------------------------------------

// X16[m][e] = (f16) emb[tokens[m]][e]
__global__ void gather_cast_kernel(const int* __restrict__ tokens,
                                   const float* __restrict__ emb,
                                   _Float16* __restrict__ X16,
                                   int E, long total) {
  long i = (long)blockIdx.x * blockDim.x + threadIdx.x;
  if (i >= total) return;
  const long m = i / E;
  const int  e = (int)(i - m * E);
  X16[i] = (_Float16)emb[(size_t)tokens[m] * E + e];
}

// WT[n][k] = (f16) W[k][n]   (W is Kd x Nd row-major)
__global__ void convt_kernel(const float* __restrict__ W,
                             _Float16* __restrict__ WT,
                             int Kd, int Nd) {
  long i = (long)blockIdx.x * blockDim.x + threadIdx.x;
  if (i >= (long)Kd * Nd) return;
  const int k = (int)(i / Nd);
  const int n = (int)(i - (long)k * Nd);
  WT[(size_t)n * Kd + k] = (_Float16)W[i];
}

__global__ void zero16_kernel(_Float16* __restrict__ p, int n) {
  int i = blockIdx.x * blockDim.x + threadIdx.x;
  if (i < n) p[i] = (_Float16)0.f;
}

// --- launch ----------------------------------------------------------------

extern "C" void kernel_launch(void* const* d_in, const int* in_sizes, int n_in,
                              void* d_out, int out_size, void* d_ws, size_t ws_size,
                              hipStream_t stream) {
  constexpr int S = 512, B = 64, E = 512, H = 1024;

  const int*   tokens = (const int*)  d_in[0];
  const float* emb    = (const float*)d_in[1];
  const float* W_ih0  = (const float*)d_in[2];
  const float* W_hh0  = (const float*)d_in[3];
  const float* b_ih0  = (const float*)d_in[4];
  const float* b_hh0  = (const float*)d_in[5];
  const float* W_ih1  = (const float*)d_in[6];
  const float* W_hh1  = (const float*)d_in[7];
  const float* b_ih1  = (const float*)d_in[8];
  const float* b_hh1  = (const float*)d_in[9];
  float* out = (float*)d_out;   // [S*B*H] y1 then [2*B*H] hidden

  char* ws = (char*)d_ws;
  size_t off = 0;
  auto carve = [&](size_t bytes) { size_t o = off; off += (bytes + 255) & ~(size_t)255; return o; };

  _Float16* X16    = (_Float16*)(ws + carve((size_t)S * B * E * 2));
  _Float16* Wih0T  = (_Float16*)(ws + carve((size_t)E * H * 2));
  _Float16* Whh0T  = (_Float16*)(ws + carve((size_t)H * H * 2));
  _Float16* Wih1T  = (_Float16*)(ws + carve((size_t)H * H * 2));
  _Float16* Whh1T  = (_Float16*)(ws + carve((size_t)H * H * 2));
  float*    xw     = (float*)   (ws + carve((size_t)S * B * H * 4)); // reused L0 then L1
  _Float16* y016   = (_Float16*)(ws + carve((size_t)S * B * H * 2));
  _Float16* hz     = (_Float16*)(ws + carve((size_t)B * H * 2));     // zero initial h
  _Float16* hbuf   = (_Float16*)(ws + carve((size_t)2 * B * H * 2)); // L1 h ping-pong
  (void)ws_size; (void)in_sizes; (void)n_in; (void)out_size;

  // Prep: weight convert+transpose, embedding gather+cast, zero h0.
  convt_kernel<<<(E * H + 255) / 256, 256, 0, stream>>>(W_ih0, Wih0T, E, H);
  convt_kernel<<<(H * H + 255) / 256, 256, 0, stream>>>(W_hh0, Whh0T, H, H);
  convt_kernel<<<(H * H + 255) / 256, 256, 0, stream>>>(W_ih1, Wih1T, H, H);
  convt_kernel<<<(H * H + 255) / 256, 256, 0, stream>>>(W_hh1, Whh1T, H, H);
  {
    long total = (long)S * B * E;
    gather_cast_kernel<<<(int)((total + 255) / 256), 256, 0, stream>>>(tokens, emb, X16, E, total);
  }
  zero16_kernel<<<(B * H + 255) / 256, 256, 0, stream>>>(hz, B * H);

  const size_t BH  = (size_t)B * H;
  const size_t SBH = (size_t)S * B * H;

  // Layer 0 input projection: xw = X @ W_ih0 + (b_ih0 + b_hh0)   [32768x512x1024]
  wmma_gemm_kernel<128, 128, H, E><<<dim3(H / 128, (S * B) / 128), 256, 0, stream>>>(
      X16, Wih0T, b_ih0, b_hh0, nullptr, xw, nullptr, nullptr, 0);

  // Layer 0 scan: h = tanh(xw[t] + h @ W_hh0), stored f16 into y0 sequence.
  for (int t = 0; t < S; ++t) {
    const _Float16* hp = (t == 0) ? hz : (y016 + (size_t)(t - 1) * BH);
    wmma_gemm_kernel<64, 128, H, H><<<dim3(H / 128, 1), 256, 0, stream>>>(
        hp, Whh0T, nullptr, nullptr, xw + (size_t)t * BH,
        nullptr, y016 + (size_t)t * BH,
        (t == S - 1) ? (out + SBH) : nullptr, 1);
  }

  // Layer 1 input projection: xw = y0 @ W_ih1 + (b_ih1 + b_hh1)  [32768x1024x1024]
  wmma_gemm_kernel<128, 128, H, H><<<dim3(H / 128, (S * B) / 128), 256, 0, stream>>>(
      y016, Wih1T, b_ih1, b_hh1, nullptr, xw, nullptr, nullptr, 0);

  // Layer 1 scan: y1[t] = h = tanh(xw[t] + h @ W_hh1); f32 to d_out, f16 ping-pong.
  for (int t = 0; t < S; ++t) {
    const _Float16* hp = (t == 0) ? hz : (hbuf + (size_t)((t - 1) & 1) * BH);
    wmma_gemm_kernel<64, 128, H, H><<<dim3(H / 128, 1), 256, 0, stream>>>(
        hp, Whh1T, nullptr, nullptr, xw + (size_t)t * BH,
        out + (size_t)t * BH, hbuf + (size_t)(t & 1) * BH,
        (t == S - 1) ? (out + SBH + BH) : nullptr, 1);
  }
}
